// CausalSelfAttention_12489764897085
// MI455X (gfx1250) — matrix operand order repaired and checked by reading, not compiled, not run on hardware
//
#include <hip/hip_runtime.h>
#include <hip/hip_bf16.h>

// ---------------- problem constants (from reference) ----------------
#define Bc   2
#define Tc   2048
#define Cc   1024
#define Hc   16
#define KVc  4
#define HDc  64
#define NREP (Hc / KVc)

typedef __bf16 bf16_t;
typedef __attribute__((ext_vector_type(16))) __bf16 bf16x16;
typedef __attribute__((ext_vector_type(8)))  float  f32x8;

__device__ __forceinline__ bf16_t to_bf16(float f) { return (bf16_t)f; }

// D = A*B + C ; A 16x32 bf16, B 32x16 bf16, C/D 16x16 f32  -> v_wmma_f32_16x16x32_bf16
__device__ __forceinline__ f32x8 wmma_bf16(bf16x16 a, bf16x16 b, f32x8 c) {
  return __builtin_amdgcn_wmma_f32_16x16x32_bf16(false, a, false, b, (short)0, c,
                                                 false, false);
}

// A-fragment (16x32, 16-bit): lane holds row (lane&15); K = k0..k0+7 and k0+16..k0+23
// with k0 = (lane>>4)*8.  row_ptr must point at (row, K=0) with K contiguous.
__device__ __forceinline__ bf16x16 load_frag_a(const bf16_t* row_ptr, int lane) {
  int k0 = (lane >> 4) * 8;
  union { bf16x16 v; uint4 q[2]; } u;
  u.q[0] = *reinterpret_cast<const uint4*>(row_ptr + k0);
  u.q[1] = *reinterpret_cast<const uint4*>(row_ptr + k0 + 16);
  return u.v;
}

// B-fragment (32x16, 16-bit): lane holds col (lane&15); K = k0..k0+7 and k0+16..k0+23,
// k0 = (lane>>4)*8.  col_ptr points at (col, K=0) of the K-contiguous (transposed) matrix.
__device__ __forceinline__ bf16x16 load_frag_b(const bf16_t* col_ptr, int lane) {
  int k0 = (lane >> 4) * 8;
  union { bf16x16 v; uint4 q[2]; } u;
  u.q[0] = *reinterpret_cast<const uint4*>(col_ptr + k0);
  u.q[1] = *reinterpret_cast<const uint4*>(col_ptr + k0 + 16);
  return u.v;
}

// CDNA5 async copy: 16 bytes global -> LDS per lane, tracked by ASYNCcnt.
// VDST = 32-bit LDS byte address (low 32 bits of generic pointer = AS3 offset),
// VADDR = 64-bit global address.  (ISA §15.18.3 op 98 GLOBAL_LOAD_ASYNC_TO_LDS_B128)
__device__ __forceinline__ void async_copy_b128(const bf16_t* g, bf16_t* l) {
  unsigned laddr = (unsigned)(size_t)l;
  unsigned long long gaddr = (unsigned long long)(size_t)g;
  asm volatile("global_load_async_to_lds_b128 %0, %1, off"
               :: "v"(laddr), "v"(gaddr)
               : "memory");
}
__device__ __forceinline__ void wait_async0() {
  asm volatile("s_wait_asynccnt 0x0" ::: "memory");
}

// ---------------- elementwise conversion kernels ----------------
__global__ void cvt_bf16_kernel(const float* __restrict__ in, bf16_t* __restrict__ out, int n) {
  int i = blockIdx.x * blockDim.x + threadIdx.x;
  if (i < n) out[i] = to_bf16(in[i]);
}

// W (K x N, f32 row-major) -> Wt (N x K, bf16 row-major)
__global__ void wtrans_kernel(const float* __restrict__ W, bf16_t* __restrict__ Wt,
                              int K, int N) {
  int i = blockIdx.x * blockDim.x + threadIdx.x;
  if (i >= K * N) return;
  int k = i % K, n = i / K;
  Wt[i] = to_bf16(W[(size_t)k * N + n]);
}

// v_f32 [B*T][KV*64] -> vT_bf [B][KV][HD][T]  (keys contiguous for P*V B-fragments)
__global__ void vtrans_kernel(const float* __restrict__ v, bf16_t* __restrict__ vt) {
  int i = blockIdx.x * blockDim.x + threadIdx.x;
  if (i >= Bc * Tc * KVc * HDc) return;
  int hd = i & 63;
  int kv = (i >> 6) & (KVc - 1);
  int t  = (i >> 8) % Tc;
  int b  = i / (Tc * KVc * HDc);
  vt[(((size_t)b * KVc + kv) * HDc + hd) * Tc + t] = to_bf16(v[i]);
}

// ---------------- tiled WMMA GEMM: D(f32, MxN) = A(bf16 MxK) * Bt(bf16 NxK)^T ----------------
// Double-buffered LDS, tiles staged via global_load_async_to_lds_b128 (ASYNCcnt).
#define BM 128
#define BN 128
#define BK 32
__global__ __launch_bounds__(256) void gemm_bf16_f32(
    const bf16_t* __restrict__ A, const bf16_t* __restrict__ Bt,
    float* __restrict__ D, int M, int N, int K) {
  __shared__ __align__(16) bf16_t sA[2][BM][BK + 8];
  __shared__ __align__(16) bf16_t sB[2][BN][BK + 8];

  const int tid  = threadIdx.x;
  const int lane = tid & 31;
  const int wave = tid >> 5;     // 8 waves
  const int wm   = wave & 3;     // 4 waves along M  (32 rows each)
  const int wn   = wave >> 2;    // 2 waves along N  (64 cols each)
  const int m0   = blockIdx.y * BM;
  const int n0   = blockIdx.x * BN;

  const int ldRow = tid >> 1;          // 0..127
  const int ldCol = (tid & 1) * 16;    // 0 / 16
  const bf16_t* agBase = A  + (size_t)(m0 + ldRow) * K + ldCol;
  const bf16_t* bgBase = Bt + (size_t)(n0 + ldRow) * K + ldCol;

  f32x8 acc[2][4] = {};

  // stage tile kt into buffer buf (async: no VGPR round-trip)
  auto stage = [&](int buf, int kt) {
    async_copy_b128(agBase + kt, &sA[buf][ldRow][ldCol]);
    async_copy_b128(bgBase + kt, &sB[buf][ldRow][ldCol]);
  };

  const int nk = K / BK;
  stage(0, 0);
  wait_async0();
  __syncthreads();

  for (int kt = 0; kt < nk; ++kt) {
    const int buf = kt & 1;
    if (kt + 1 < nk) stage(buf ^ 1, (kt + 1) * BK);  // overlap copy with WMMA

    bf16x16 afrag[2], bfrag[4];
#pragma unroll
    for (int i = 0; i < 2; ++i)
      afrag[i] = load_frag_a(&sA[buf][wm * 32 + i * 16 + (lane & 15)][0], lane);
#pragma unroll
    for (int j = 0; j < 4; ++j)
      bfrag[j] = load_frag_b(&sB[buf][wn * 64 + j * 16 + (lane & 15)][0], lane);

#pragma unroll
    for (int i = 0; i < 2; ++i)
#pragma unroll
      for (int j = 0; j < 4; ++j)
        acc[i][j] = wmma_bf16(afrag[i], bfrag[j], acc[i][j]);

    wait_async0();     // my async fills of buf^1 are in LDS
    __syncthreads();   // everyone done reading buf and filling buf^1
  }

  const int hi = lane >> 4, ln = lane & 15;
#pragma unroll
  for (int i = 0; i < 2; ++i)
#pragma unroll
    for (int j = 0; j < 4; ++j)
#pragma unroll
      for (int v = 0; v < 8; ++v) {
        int r = m0 + wm * 32 + i * 16 + v + 8 * hi;   // C-layout row
        int c = n0 + wn * 64 + j * 16 + ln;           // C-layout col
        D[(size_t)r * N + c] = acc[i][j][v];
      }
}

// ---------------- RoPE + RMSNorm epilogue (one wave per (b,t,head)) ----------------
__global__ __launch_bounds__(256) void rope_rms_kernel(
    const float* __restrict__ proj,   // [B*T][NH*64] f32
    const float* __restrict__ cosb,   // [T][32]
    const float* __restrict__ sinb,   // [T][32]
    bf16_t* __restrict__ out,         // [B][NH][T][64] bf16
    int NH, float scale) {
  int gwave = (blockIdx.x * blockDim.x + threadIdx.x) >> 5;
  int lane  = threadIdx.x & 31;
  int h  = gwave % NH;
  int bt = gwave / NH;
  int t  = bt % Tc;
  int b  = bt / Tc;

  const float* src = proj + (size_t)bt * (NH * HDc) + h * HDc;
  float x1 = src[lane], x2 = src[lane + 32];
  float c = cosb[t * 32 + lane], s = sinb[t * 32 + lane];
  float r1 =  x1 * c + x2 * s;
  float r2 = -x1 * s + x2 * c;

  float ss = r1 * r1 + r2 * r2;
#pragma unroll
  for (int m = 16; m >= 1; m >>= 1) ss += __shfl_xor(ss, m, 32);
  float inv = rsqrtf(ss * (1.0f / HDc) + 1e-6f) * scale;

  bf16_t* dst = out + (((size_t)b * NH + h) * Tc + t) * HDc;
  dst[lane]      = to_bf16(r1 * inv);
  dst[lane + 32] = to_bf16(r2 * inv);
}

// ---------------- flash attention (causal, GQA), WMMA for QK^T and PV ----------------
#define TKB 64   // keys per block
__global__ __launch_bounds__(256) void flash_kernel(
    const bf16_t* __restrict__ Q,   // [B][H][T][64] pre-scaled by 1/sqrt(HD)
    const bf16_t* __restrict__ Kb,  // [B][KV][T][64]
    const bf16_t* __restrict__ Vt,  // [B][KV][64][T]
    bf16_t* __restrict__ Y) {       // [B][T][H][64]
  __shared__ __align__(16) bf16_t sP[8][16 * TKB];  // per-wave 16x64 P buffer

  const int lane = threadIdx.x & 31;
  const int wave = threadIdx.x >> 5;
  const int qb = blockIdx.x % (Tc / 128);
  const int h  = (blockIdx.x / (Tc / 128)) % Hc;
  const int b  = blockIdx.x / ((Tc / 128) * Hc);
  const int kvh = h / NREP;
  const int q0 = qb * 128 + wave * 16;

  const bf16_t* qp = Q  + (((size_t)b * Hc  + h)   * Tc) * HDc;
  const bf16_t* kp = Kb + (((size_t)b * KVc + kvh) * Tc) * HDc;
  const bf16_t* vp = Vt + (((size_t)b * KVc + kvh) * HDc) * Tc;
  bf16_t* pbuf = &sP[wave][0];

  const int hi = lane >> 4, ln = lane & 15;

  bf16x16 qfrag[2];
  qfrag[0] = load_frag_a(qp + (size_t)(q0 + ln) * HDc,      lane);  // hd 0..31
  qfrag[1] = load_frag_a(qp + (size_t)(q0 + ln) * HDc + 32, lane);  // hd 32..63

  f32x8 oacc[4] = {};
  float mrow[8], lrow[8];
#pragma unroll
  for (int v = 0; v < 8; ++v) { mrow[v] = -1e30f; lrow[v] = 0.0f; }

  const int kend = q0 + 16;  // causal: keys < q0+16
  for (int kb = 0; kb < kend; kb += TKB) {
    // ---- S = Q K^T over 64 keys (4 N-frags x 2 K-steps) ----
    f32x8 sacc[4] = {};
#pragma unroll
    for (int nf = 0; nf < 4; ++nf) {
      const bf16_t* kcol = kp + (size_t)(kb + nf * 16 + ln) * HDc;
      sacc[nf] = wmma_bf16(qfrag[0], load_frag_b(kcol,      lane), sacc[nf]);
      sacc[nf] = wmma_bf16(qfrag[1], load_frag_b(kcol + 32, lane), sacc[nf]);
    }
    // ---- causal mask on diagonal blocks ----
    if (kb + TKB > q0) {
#pragma unroll
      for (int nf = 0; nf < 4; ++nf)
#pragma unroll
        for (int v = 0; v < 8; ++v) {
          int r = q0 + v + 8 * hi;
          int c = kb + nf * 16 + ln;
          if (c > r) sacc[nf][v] = -1e30f;
        }
    }
    // ---- online softmax: row max across the 16 lanes that share a row ----
    float mnew[8], fac[8];
#pragma unroll
    for (int v = 0; v < 8; ++v) {
      float m = fmaxf(fmaxf(sacc[0][v], sacc[1][v]), fmaxf(sacc[2][v], sacc[3][v]));
#pragma unroll
      for (int sh = 8; sh >= 1; sh >>= 1) m = fmaxf(m, __shfl_xor(m, sh, 16));
      mnew[v] = fmaxf(mrow[v], m);
      fac[v]  = __expf(mrow[v] - mnew[v]);
      mrow[v] = mnew[v];
    }
    // ---- P = exp(S - m): row sums + stash P in LDS (C-layout -> row-major) ----
    float bsum[8] = {};
#pragma unroll
    for (int nf = 0; nf < 4; ++nf)
#pragma unroll
      for (int v = 0; v < 8; ++v) {
        float p = __expf(sacc[nf][v] - mnew[v]);
        bsum[v] += p;
        pbuf[(v + 8 * hi) * TKB + nf * 16 + ln] = to_bf16(p);
      }
#pragma unroll
    for (int v = 0; v < 8; ++v) {
      float s2 = bsum[v];
#pragma unroll
      for (int sh = 8; sh >= 1; sh >>= 1) s2 += __shfl_xor(s2, sh, 16);
      lrow[v] = lrow[v] * fac[v] + s2;
    }
#pragma unroll
    for (int j = 0; j < 4; ++j)
#pragma unroll
      for (int v = 0; v < 8; ++v) oacc[j][v] *= fac[v];

    asm volatile("s_wait_dscnt 0x0" ::: "memory");  // P stores -> P reloads (same wave)

    // ---- O += P V : A-frags from LDS, B-frags from transposed V ----
    bf16x16 pf0 = load_frag_a(pbuf + ln * TKB,      lane);  // keys 0..31
    bf16x16 pf1 = load_frag_a(pbuf + ln * TKB + 32, lane);  // keys 32..63
#pragma unroll
    for (int j = 0; j < 4; ++j) {
      const bf16_t* vcol = vp + (size_t)(j * 16 + ln) * Tc + kb;
      oacc[j] = wmma_bf16(pf0, load_frag_b(vcol,      lane), oacc[j]);
      oacc[j] = wmma_bf16(pf1, load_frag_b(vcol + 32, lane), oacc[j]);
    }
  }
  // ---- epilogue: O / l -> Y[b][t][h][hd] ----
#pragma unroll
  for (int v = 0; v < 8; ++v) {
    float invl = 1.0f / lrow[v];
    int t = q0 + v + 8 * hi;
    bf16_t* yrow = Y + (((size_t)b * Tc + t) * Hc + h) * HDc;
#pragma unroll
    for (int j = 0; j < 4; ++j)
      yrow[j * 16 + ln] = to_bf16(oacc[j][v] * invl);
  }
}

// ---------------- host-side orchestration ----------------
extern "C" void kernel_launch(void* const* d_in, const int* in_sizes, int n_in,
                              void* d_out, int out_size, void* d_ws, size_t ws_size,
                              hipStream_t stream) {
  (void)in_sizes; (void)n_in; (void)out_size; (void)ws_size;
  const float* x   = (const float*)d_in[0];
  const float* cs  = (const float*)d_in[1];
  const float* sn  = (const float*)d_in[2];
  const float* Wq  = (const float*)d_in[3];
  const float* Wk  = (const float*)d_in[4];
  const float* Wv  = (const float*)d_in[5];
  const float* Wo  = (const float*)d_in[6];
  float* out = (float*)d_out;

  const size_t MT = (size_t)Bc * Tc;         // 4096 rows
  char* ws = (char*)d_ws;
  size_t off = 0;
  auto take = [&](size_t bytes) { char* p = ws + off; off += (bytes + 255) & ~(size_t)255; return p; };

  bf16_t* x_bf  = (bf16_t*)take(MT * Cc * 2);                 // x in bf16
  bf16_t* Wqt   = (bf16_t*)take((size_t)Cc * Cc * 2);         // Wq^T (N x K)
  bf16_t* Wkt   = (bf16_t*)take((size_t)(KVc * HDc) * Cc * 2);
  bf16_t* Wvt   = (bf16_t*)take((size_t)(KVc * HDc) * Cc * 2);
  bf16_t* Wot   = (bf16_t*)take((size_t)Cc * Cc * 2);         // Wo^T
  float*  qf    = (float*)take(MT * Cc * 4);                  // q proj f32 (reused for y later)
  float*  kf    = (float*)take(MT * (KVc * HDc) * 4);
  float*  vf    = (float*)take(MT * (KVc * HDc) * 4);
  bf16_t* q_bf  = (bf16_t*)take((size_t)Bc * Hc * Tc * HDc * 2);   // [B][H][T][HD]
  bf16_t* k_bf  = (bf16_t*)take((size_t)Bc * KVc * Tc * HDc * 2);  // [B][KV][T][HD]
  bf16_t* vT_bf = (bf16_t*)take((size_t)Bc * KVc * HDc * Tc * 2);  // [B][KV][HD][T]
  bf16_t* y_bf  = (bf16_t*)qf;   // alias: qf free after rope epilogue

  // 1) convert x, transpose+convert weights
  {
    int n = (int)(MT * Cc);
    cvt_bf16_kernel<<<(n + 255) / 256, 256, 0, stream>>>(x, x_bf, n);
  }
  wtrans_kernel<<<(Cc * Cc + 255) / 256, 256, 0, stream>>>(Wq, Wqt, Cc, Cc);
  wtrans_kernel<<<(Cc * KVc * HDc + 255) / 256, 256, 0, stream>>>(Wk, Wkt, Cc, KVc * HDc);
  wtrans_kernel<<<(Cc * KVc * HDc + 255) / 256, 256, 0, stream>>>(Wv, Wvt, Cc, KVc * HDc);
  wtrans_kernel<<<(Cc * Cc + 255) / 256, 256, 0, stream>>>(Wo, Wot, Cc, Cc);

  // 2) QKV projections (WMMA GEMMs, async-LDS staged)
  gemm_bf16_f32<<<dim3(Cc / BN, MT / BM), 256, 0, stream>>>(x_bf, Wqt, qf, (int)MT, Cc, Cc);
  gemm_bf16_f32<<<dim3((KVc * HDc) / BN, MT / BM), 256, 0, stream>>>(x_bf, Wkt, kf, (int)MT, KVc * HDc, Cc);
  gemm_bf16_f32<<<dim3((KVc * HDc) / BN, MT / BM), 256, 0, stream>>>(x_bf, Wvt, vf, (int)MT, KVc * HDc, Cc);

  // 3) RoPE + RMSNorm epilogues (scale 1/sqrt(HD) folded into Q), V transpose
  {
    int qwaves = Bc * Tc * Hc;
    rope_rms_kernel<<<qwaves / 8, 256, 0, stream>>>(qf, cs, sn, q_bf, Hc, 0.125f);
    int kwaves = Bc * Tc * KVc;
    rope_rms_kernel<<<kwaves / 8, 256, 0, stream>>>(kf, cs, sn, k_bf, KVc, 1.0f);
    int n = Bc * Tc * KVc * HDc;
    vtrans_kernel<<<(n + 255) / 256, 256, 0, stream>>>(vf, vT_bf);
  }

  // 4) causal flash attention (GQA)
  flash_kernel<<<Bc * Hc * (Tc / 128), 256, 0, stream>>>(q_bf, k_bf, vT_bf, y_bf);

  // 5) output projection -> f32 result
  gemm_bf16_f32<<<dim3(Cc / BN, MT / BM), 256, 0, stream>>>(y_bf, Wot, out, (int)MT, Cc, Cc);
}
